// GroupedQueryAttention_36696200577421
// MI455X (gfx1250) — compile-verified, matrix operand order checked by reading, not run on hardware
//
#include <hip/hip_runtime.h>
#include <hip/hip_bf16.h>

typedef __attribute__((ext_vector_type(16))) __bf16 v16bf;
typedef __attribute__((ext_vector_type(8)))  float  v8f;
typedef __attribute__((ext_vector_type(8)))  __bf16 v8bf;
typedef __attribute__((ext_vector_type(4)))  __bf16 v4bf;
typedef __attribute__((ext_vector_type(4)))  int    v4i;

#define B_SZ 64
#define T_SZ 257
#define DM   1024
#define NH   16
#define NG   4
#define HD   64
#define MROWS (B_SZ * T_SZ)          /* 16448 */
#define QKV_N (NH*HD + 2*NG*HD)      /* 1536  */

// Async global->LDS copies (CDNA5 GLOBAL_LOAD_ASYNC_TO_LDS_B128, ASYNCcnt).
// Guarded so the file compiles on toolchains without the builtin (and on the
// host pass, where target builtins are invisible).
#if defined(__gfx1250__) && __has_builtin(__builtin_amdgcn_global_load_async_to_lds_b128)
#define USE_ASYNC_LDS 1
#else
#define USE_ASYNC_LDS 0
#endif

#define GLOBAL_AS __attribute__((address_space(1)))
#define LDS_AS    __attribute__((address_space(3)))

#if USE_ASYNC_LDS
__device__ __forceinline__ void async_wait0() {
#if __has_builtin(__builtin_amdgcn_s_wait_asynccnt)
  __builtin_amdgcn_s_wait_asynccnt(0);
#else
  asm volatile("s_wait_asynccnt 0" ::: "memory");
#endif
}
// 16-byte async copy: global (AS1) -> LDS (AS3), tracked by ASYNCcnt.
__device__ __forceinline__ void async_cp16(const __bf16* gsrc, __bf16* ldst) {
  __builtin_amdgcn_global_load_async_to_lds_b128(
      (GLOBAL_AS v4i*)gsrc, (LDS_AS v4i*)ldst, 0, 0);
}
#endif

// ---------------------------------------------------------------------------
// WMMA fragment loaders per CDNA5 ISA 7.12.2 (wave32).
// A (16x32 bf16): lane L holds row M=L%16; elem e -> K = e + 8*(e>=8) + 8*(L>=16)
// B (32x16 bf16): lane L holds col N=L%16; same K mapping, read K-major.
// C/D (16x16 f32): lane L, VGPR r -> M = r + 8*(L>=16), N = L%16
// ---------------------------------------------------------------------------
__device__ __forceinline__ v16bf load_a_frag(const __bf16* p, int ld) {
  const int lane = threadIdx.x & 31;
  const int m    = lane & 15;
  const int koff = (lane >> 4) << 3;
  v16bf a;
#pragma unroll
  for (int e = 0; e < 16; ++e) {
    const int k = e + ((e >> 3) << 3) + koff;
    a[e] = p[m * ld + k];
  }
  return a;
}

__device__ __forceinline__ v16bf load_b_frag(const __bf16* p, int ld) {
  const int lane = threadIdx.x & 31;
  const int n    = lane & 15;
  const int koff = (lane >> 4) << 3;
  v16bf b;
#pragma unroll
  for (int e = 0; e < 16; ++e) {
    const int k = e + ((e >> 3) << 3) + koff;
    b[e] = p[k * ld + n];
  }
  return b;
}

__device__ __forceinline__ void st_out(float*  p, float v) { *p = v; }
__device__ __forceinline__ void st_out(__bf16* p, float v) { *p = (__bf16)v; }

// ---------------------------------------------------------------------------
// Generic GEMM: C[MxN] = A[MxK] * B[KxN], bf16 WMMA, fp32 accumulate.
// 256 threads = 8 waves; block tile 128x128; wave tile 32x64; K-step 32.
// Software pipelined: tile k+1 global->reg loads issued before tile k WMMAs.
// ---------------------------------------------------------------------------
template <typename TA, typename TC>
__global__ __launch_bounds__(256) void gemm_wmma(const TA* __restrict__ A,
                                                 const float* __restrict__ Bw,
                                                 TC* __restrict__ C,
                                                 int M, int N, int K) {
  __shared__ __bf16 As[128][40];    // 128 rows x 32 K  (pad 40)
  __shared__ __bf16 Bs[32][136];    // 32 K  x 128 N   (pad 136)
  const int tid  = threadIdx.x;
  const int lane = tid & 31;
  const int wave = tid >> 5;
  const int wm   = (wave & 3) * 32;   // 4 waves along M
  const int wn   = (wave >> 2) * 64;  // 2 waves along N
  const int rowBase = blockIdx.x * 128;
  const int colBase = blockIdx.y * 128;

  float aR[4][4], bR[4][4];

  auto loadA = [&](int k0) {
#pragma unroll
    for (int c = 0; c < 4; ++c) {
      const int idx = tid + c * 256;          // 1024 chunks of 4
      const int r = idx >> 3, k4 = (idx & 7) << 2;
      const int row = rowBase + r;
      if (row < M) {
        if constexpr (sizeof(TA) == 4) {
          const float4 v = *(const float4*)&A[(size_t)row * K + k0 + k4];
          aR[c][0] = v.x; aR[c][1] = v.y; aR[c][2] = v.z; aR[c][3] = v.w;
        } else {
          const v4bf v = *(const v4bf*)&A[(size_t)row * K + k0 + k4];
          aR[c][0] = (float)v[0]; aR[c][1] = (float)v[1];
          aR[c][2] = (float)v[2]; aR[c][3] = (float)v[3];
        }
      } else {
        aR[c][0] = aR[c][1] = aR[c][2] = aR[c][3] = 0.0f;
      }
    }
  };
  auto loadB = [&](int k0) {
#pragma unroll
    for (int c = 0; c < 4; ++c) {
      const int idx = tid + c * 256;          // 1024 chunks of 4
      const int kk = idx >> 5, n4 = (idx & 31) << 2;
      const float4 v = *(const float4*)&Bw[(size_t)(k0 + kk) * N + colBase + n4];
      bR[c][0] = v.x; bR[c][1] = v.y; bR[c][2] = v.z; bR[c][3] = v.w;
    }
  };
  auto stage = [&]() {
#pragma unroll
    for (int c = 0; c < 4; ++c) {
      const int idx = tid + c * 256;
      {
        const int r = idx >> 3, k4 = (idx & 7) << 2;
        v4bf h;
        h[0] = (__bf16)aR[c][0]; h[1] = (__bf16)aR[c][1];
        h[2] = (__bf16)aR[c][2]; h[3] = (__bf16)aR[c][3];
        *(v4bf*)&As[r][k4] = h;
      }
      {
        const int kk = idx >> 5, n4 = (idx & 31) << 2;
        v4bf h;
        h[0] = (__bf16)bR[c][0]; h[1] = (__bf16)bR[c][1];
        h[2] = (__bf16)bR[c][2]; h[3] = (__bf16)bR[c][3];
        *(v4bf*)&Bs[kk][n4] = h;
      }
    }
  };

  v8f acc[2][4] = {};
  loadA(0);
  loadB(0);

  for (int k0 = 0; k0 < K; k0 += 32) {
    stage();
    __syncthreads();

    // Issue next tile's global loads before this tile's WMMAs (latency overlap).
    if (k0 + 32 < K) {
      loadA(k0 + 32);
      loadB(k0 + 32);
    }
    // L2 prefetch two K-steps ahead (global_prefetch_b8).
    if (k0 + 64 < K) {
      const int pr = rowBase + (tid >> 3);
      __builtin_prefetch(&A[(size_t)(pr < M ? pr : M - 1) * K + k0 + 64], 0, 1);
      __builtin_prefetch(&Bw[(size_t)(k0 + 64 + (tid >> 5)) * N + colBase + ((tid & 31) << 2)], 0, 1);
    }

    const v16bf a0 = load_a_frag(&As[wm][0], 40);
    const v16bf a1 = load_a_frag(&As[wm + 16][0], 40);
#pragma unroll
    for (int j = 0; j < 4; ++j) {
      const v16bf b = load_b_frag(&Bs[0][wn + j * 16], 136);
      acc[0][j] = __builtin_amdgcn_wmma_f32_16x16x32_bf16(false, a0, false, b, (short)0, acc[0][j], false, false);
      acc[1][j] = __builtin_amdgcn_wmma_f32_16x16x32_bf16(false, a1, false, b, (short)0, acc[1][j], false, false);
    }
    __syncthreads();
  }

  const int half = lane >> 4, nn = lane & 15;
#pragma unroll
  for (int i = 0; i < 2; ++i)
#pragma unroll
    for (int j = 0; j < 4; ++j)
#pragma unroll
      for (int r = 0; r < 8; ++r) {
        const int row = rowBase + wm + i * 16 + r + 8 * half;
        const int col = colBase + wn + j * 16 + nn;
        if (row < M) st_out(&C[(size_t)row * N + col], acc[i][j][r]);
      }
}

// ---------------------------------------------------------------------------
// 2-D RoPE + head split/transpose. slot: 0..15 = Q heads (scaled by 1/sqrt(hd)),
// 16..19 = K groups (rotated), 20..23 = V groups (copy).
// ---------------------------------------------------------------------------
__global__ __launch_bounds__(256) void rope_split(const __bf16* __restrict__ qkv,
                                                  __bf16* __restrict__ Qb,
                                                  __bf16* __restrict__ Kb,
                                                  __bf16* __restrict__ Vb) {
  const int idx = blockIdx.x * blockDim.x + threadIdx.x;
  const int total = B_SZ * 24 * T_SZ * 16;
  if (idx >= total) return;
  const int f    = idx & 15;
  const int t    = (idx >> 4) % T_SZ;
  const int slot = (idx / (16 * T_SZ)) % 24;
  const int b    = idx / (16 * T_SZ * 24);

  int colbase;
  if (slot < 16)      colbase = slot * HD;
  else if (slot < 20) colbase = NH * HD + (slot - 16) * HD;
  else                colbase = NH * HD + NG * HD + (slot - 20) * HD;

  const __bf16* src = qkv + ((size_t)(b * T_SZ + t)) * QKV_N + colbase;
  const float x1 = (float)src[f],      x2 = (float)src[f + 16];
  const float y1 = (float)src[f + 32], y2 = (float)src[f + 48];
  float o0 = x1, o1 = x2, o2 = y1, o3 = y2;

  if (slot < 20 && t > 0) {             // rotate Q/K patch tokens; cls unchanged
    const int p = t - 1;
    const float gx = (float)(p >> 4);   // 16x16 patch grid, 'ij' order
    const float gy = (float)(p & 15);
    // inv_freq = 10000^(-f/16) = 2^(-f * log2(10000)/16)
    const float inv = exp2f(-(float)f * (13.287712379549449f / 16.0f));
    const float thx = gx * inv, thy = gy * inv;
    const float sx = __sinf(thx), cx = __cosf(thx);
    const float sy = __sinf(thy), cy = __cosf(thy);
    o0 = x1 * cx - x2 * sx;
    o1 = x1 * sx + x2 * cx;
    o2 = y1 * cy - y2 * sy;
    o3 = y1 * sy + y2 * cy;
  }

  __bf16* dst;
  if (slot < 16) {
    // Fold softmax scale 1/sqrt(64) into Q.
    o0 *= 0.125f; o1 *= 0.125f; o2 *= 0.125f; o3 *= 0.125f;
    dst = Qb + (((size_t)(b * NH + slot)) * T_SZ + t) * HD;
  } else if (slot < 20) {
    dst = Kb + (((size_t)(b * NG + (slot - 16))) * T_SZ + t) * HD;
  } else {
    dst = Vb + (((size_t)(b * NG + (slot - 20))) * T_SZ + t) * HD;
  }
  dst[f]      = (__bf16)o0;
  dst[f + 16] = (__bf16)o1;
  dst[f + 32] = (__bf16)o2;
  dst[f + 48] = (__bf16)o3;
}

// ---------------------------------------------------------------------------
// Flash attention, one (b,h) per blockIdx.y. 256 threads = 8 waves; each wave
// owns one 16-query tile (17 tiles -> gridDim.x = 3 groups of 8). Online
// softmax over 32-key tiles staged in LDS via async global->LDS copies.
// ---------------------------------------------------------------------------
__global__ __launch_bounds__(256) void attn_wmma(const __bf16* __restrict__ Qb,
                                                 const __bf16* __restrict__ Kb,
                                                 const __bf16* __restrict__ Vb,
                                                 __bf16* __restrict__ Ob) {
  __shared__ __bf16 Qs[8][16][72];
  __shared__ __bf16 Ks[32][72];
  __shared__ __bf16 Vs[32][72];
  __shared__ __bf16 Ps[8][16][40];

  const int tid  = threadIdx.x;
  const int lane = tid & 31;
  const int wave = tid >> 5;
  const int bh = blockIdx.y;
  const int b = bh >> 4, h = bh & 15, g = h >> 2;
  const size_t qoff  = (size_t)bh * T_SZ * HD;
  const size_t kvoff = (size_t)(b * NG + g) * T_SZ * HD;

  // Stage the 8 Q tiles (128 rows x 64 dims); zero-pad rows >= T.
#pragma unroll
  for (int i = 0; i < 4; ++i) {
    const int c = tid + i * 256;          // 1024 chunks of 8 bf16
    const int wv = c >> 7;
    const int row = (c >> 3) & 15;
    const int col8 = (c & 7) << 3;
    const int t = (blockIdx.x * 8 + wv) * 16 + row;
#if USE_ASYNC_LDS
    if (t < T_SZ) {
      async_cp16(&Qb[qoff + (size_t)t * HD + col8], &Qs[wv][row][col8]);
    } else {
      const v8bf z = {};
      *(v8bf*)&Qs[wv][row][col8] = z;   // ds path; disjoint from async writes
    }
#else
    v8bf val = {};
    if (t < T_SZ) val = *(const v8bf*)&Qb[qoff + (size_t)t * HD + col8];
    *(v8bf*)&Qs[wv][row][col8] = val;
#endif
  }
#if USE_ASYNC_LDS
  async_wait0();
#endif
  __syncthreads();

  const int qbase = (blockIdx.x * 8 + wave) * 16;
  const v16bf qa0 = load_a_frag(&Qs[wave][0][0], 72);
  const v16bf qa1 = load_a_frag(&Qs[wave][0][32], 72);

  v8f o[4] = {{}, {}, {}, {}};
  float mrow[8], lrow[8];
#pragma unroll
  for (int r = 0; r < 8; ++r) { mrow[r] = -1e30f; lrow[r] = 0.0f; }

  const int NKT = (T_SZ + 31) / 32;  // 9 key tiles; only the last is partial
  for (int kt = 0; kt < NKT; ++kt) {
    __syncthreads();
    // Stage K and V tiles (32 keys x 64 dims). Keys >= T keep stale (finite)
    // LDS data: their scores are masked and their probabilities are exactly 0.
    {
      const int row = tid >> 3;            // 256 chunks of 8 each
      const int col8 = (tid & 7) << 3;
      const int key = kt * 32 + row;
      if (key < T_SZ) {
#if USE_ASYNC_LDS
        async_cp16(&Kb[kvoff + (size_t)key * HD + col8], &Ks[row][col8]);
        async_cp16(&Vb[kvoff + (size_t)key * HD + col8], &Vs[row][col8]);
#else
        *(v8bf*)&Ks[row][col8] = *(const v8bf*)&Kb[kvoff + (size_t)key * HD + col8];
        *(v8bf*)&Vs[row][col8] = *(const v8bf*)&Vb[kvoff + (size_t)key * HD + col8];
#endif
      }
    }
#if USE_ASYNC_LDS
    async_wait0();
#endif
    __syncthreads();

    // S = Q * K^T : K rows act as the K-major B operand (A-frag pattern).
    v8f s0 = {}, s1 = {};
    {
      v16bf bk = load_a_frag(&Ks[0][0], 72);
      s0 = __builtin_amdgcn_wmma_f32_16x16x32_bf16(false, qa0, false, bk, (short)0, s0, false, false);
      bk = load_a_frag(&Ks[0][32], 72);
      s0 = __builtin_amdgcn_wmma_f32_16x16x32_bf16(false, qa1, false, bk, (short)0, s0, false, false);
      bk = load_a_frag(&Ks[16][0], 72);
      s1 = __builtin_amdgcn_wmma_f32_16x16x32_bf16(false, qa0, false, bk, (short)0, s1, false, false);
      bk = load_a_frag(&Ks[16][32], 72);
      s1 = __builtin_amdgcn_wmma_f32_16x16x32_bf16(false, qa1, false, bk, (short)0, s1, false, false);
    }

    // Masking needed only on the final (partial) key tile.
    const bool lastTile = (kt == NKT - 1);
    const bool v0ok = !lastTile || (kt * 32 + (lane & 15)) < T_SZ;
    const bool v1ok = !lastTile || (kt * 32 + 16 + (lane & 15)) < T_SZ;
    const int prow = 8 * (lane >> 4);
#pragma unroll
    for (int r = 0; r < 8; ++r) {
      const float x0 = v0ok ? s0[r] : -1e30f;
      const float x1 = v1ok ? s1[r] : -1e30f;
      float mr = fmaxf(x0, x1);
#pragma unroll
      for (int off = 1; off < 16; off <<= 1) mr = fmaxf(mr, __shfl_xor(mr, off, 32));
      const float mnew  = fmaxf(mrow[r], mr);
      const float alpha = __expf(mrow[r] - mnew);
      const float p0 = __expf(x0 - mnew);
      const float p1 = __expf(x1 - mnew);
      float sr = p0 + p1;
#pragma unroll
      for (int off = 1; off < 16; off <<= 1) sr += __shfl_xor(sr, off, 32);
      lrow[r] = lrow[r] * alpha + sr;
      mrow[r] = mnew;
#pragma unroll
      for (int nt = 0; nt < 4; ++nt) o[nt][r] *= alpha;
      // D-layout -> LDS so P can be re-read as an A fragment.
      Ps[wave][r + prow][lane & 15]        = (__bf16)p0;
      Ps[wave][r + prow][16 + (lane & 15)] = (__bf16)p1;
    }
    __syncthreads();

    // O += P (16x32) * V (32x64)
    const v16bf pa = load_a_frag(&Ps[wave][0][0], 40);
#pragma unroll
    for (int nt = 0; nt < 4; ++nt) {
      const v16bf bv = load_b_frag(&Vs[0][nt * 16], 72);
      o[nt] = __builtin_amdgcn_wmma_f32_16x16x32_bf16(false, pa, false, bv, (short)0, o[nt], false, false);
    }
  }

  // Epilogue: normalize and scatter into [b, t, h*64 + d] bf16.
  const int half = lane >> 4, nn = lane & 15;
#pragma unroll
  for (int r = 0; r < 8; ++r) {
    const int t = qbase + r + 8 * half;
    if (t < T_SZ) {
      const float inv_l = 1.0f / lrow[r];
#pragma unroll
      for (int nt = 0; nt < 4; ++nt) {
        Ob[((size_t)(b * T_SZ + t)) * DM + h * HD + nt * 16 + nn] =
            (__bf16)(o[nt][r] * inv_l);
      }
    }
  }
}

// ---------------------------------------------------------------------------
extern "C" void kernel_launch(void* const* d_in, const int* in_sizes, int n_in,
                              void* d_out, int out_size, void* d_ws, size_t ws_size,
                              hipStream_t stream) {
  (void)in_sizes; (void)n_in; (void)out_size; (void)ws_size;
  const float* x     = (const float*)d_in[0];
  const float* w_qkv = (const float*)d_in[1];
  const float* w_o   = (const float*)d_in[2];
  // d_in[3] = window_size = [-1,-1] -> full attention, unused.

  char* ws = (char*)d_ws;
  size_t off = 0;
  auto carve = [&](size_t bytes) -> void* {
    void* p = ws + off;
    off += (bytes + 255) & ~(size_t)255;
    return p;
  };
  __bf16* qkv = (__bf16*)carve((size_t)MROWS * QKV_N * sizeof(__bf16));  // 50.5 MB
  __bf16* Qb  = (__bf16*)carve((size_t)B_SZ * NH * T_SZ * HD * sizeof(__bf16));
  __bf16* Kb  = (__bf16*)carve((size_t)B_SZ * NG * T_SZ * HD * sizeof(__bf16));
  __bf16* Vb  = (__bf16*)carve((size_t)B_SZ * NG * T_SZ * HD * sizeof(__bf16));
  // Attention output reuses the qkv buffer (qkv fully consumed by rope_split).
  __bf16* attn = qkv;
  float* out = (float*)d_out;

  // 1) qkv = x @ w_qkv   (16448 x 1536, K=1024)
  gemm_wmma<float, __bf16>
      <<<dim3((MROWS + 127) / 128, QKV_N / 128), 256, 0, stream>>>(
          x, w_qkv, qkv, MROWS, QKV_N, DM);

  // 2) RoPE + split/transpose into per-head bf16 layouts (Q pre-scaled).
  const int total = B_SZ * 24 * T_SZ * 16;
  rope_split<<<(total + 255) / 256, 256, 0, stream>>>(qkv, Qb, Kb, Vb);

  // 3) Flash attention (17 query tiles -> 3 groups of 8 waves).
  attn_wmma<<<dim3(3, B_SZ * NH), 256, 0, stream>>>(Qb, Kb, Vb, attn);

  // 4) out = attn @ w_o  (16448 x 1024, K=1024), fp32 output.
  gemm_wmma<__bf16, float>
      <<<dim3((MROWS + 127) / 128, DM / 128), 256, 0, stream>>>(
          attn, w_o, out, MROWS, DM, DM);
}
